// BezConv_16870631539247
// MI455X (gfx1250) — compile-verified, matrix operand order checked by reading, not run on hardware
//
#include <hip/hip_runtime.h>
#include <cstdint>
#include <cstddef>

// ---------------------------------------------------------------------------
// Types for CDNA5 WMMA
// ---------------------------------------------------------------------------
typedef __attribute__((ext_vector_type(16))) __bf16 v16bf;
typedef __attribute__((ext_vector_type(8)))  float  v8f;

__device__ __forceinline__ unsigned short f2bf(float f) {
  unsigned int u = __float_as_uint(f);
  unsigned int r = u + 0x7FFFu + ((u >> 16) & 1u);   // round-to-nearest-even
  return (unsigned short)(r >> 16);
}

__device__ __forceinline__ int clampi(int v, int lo, int hi) {
  return v < lo ? lo : (v > hi ? hi : v);
}

// ---------------------------------------------------------------------------
// WMMA GEMM:  C[M][Np] = A[M][Kp] * Bt[N][Kp]^T  (+ bias[n/biasDiv])
// A, Bt bf16, K contiguous. Kp % 32 == 0, Np % 32 == 0, M % 32 == 0.
// Each wave computes a 32x32 C block as 2x2 WMMA tiles: 2 A frags + 2 B frags
// feed 4 back-to-back v_wmma_f32_16x16x32_bf16 per K-step (2x operand reuse
// vs one-tile-per-wave, 4 loads/WMMA instead of 8).
// Fragment packing follows the CDNA5 16-bit A/B layout:
//   VGPR q, lane l -> k0 = (q>=4?16:0) + (l>=16?8:0) + 2*(q&3).
// ---------------------------------------------------------------------------
__global__ void __launch_bounds__(256)
wmma_gemm_bf16(const unsigned short* __restrict__ A,
               const unsigned short* __restrict__ Bt,
               const float* __restrict__ bias, int biasDiv,
               float* __restrict__ C, int M, int Nreal, int Np, int Kp)
{
  int mBlocks = M >> 5;
  int nBlocks = Np >> 5;
  int blk = blockIdx.x * blockDim.y + threadIdx.y;      // wave-uniform
  if (blk >= mBlocks * nBlocks) return;                 // whole wave exits
  int mt = (blk % mBlocks) << 5;
  int nt = (blk / mBlocks) << 5;

  int lane = threadIdx.x;
  int lm = lane & 15;
  int lh = lane >> 4;
  int kb = lh << 3;

  const unsigned short* Arow0 = A  + (size_t)(mt + lm)      * Kp;
  const unsigned short* Arow1 = A  + (size_t)(mt + 16 + lm) * Kp;
  const unsigned short* Brow0 = Bt + (size_t)(nt + lm)      * Kp;
  const unsigned short* Brow1 = Bt + (size_t)(nt + 16 + lm) * Kp;

  union Frag { unsigned int u[8]; v16bf v; };
  Frag a0, a1, b0, b1;
  v8f acc00 = {}, acc01 = {}, acc10 = {}, acc11 = {};

  for (int k = 0; k < Kp; k += 32) {
    __builtin_prefetch(Arow0 + k + 32, 0, 1);
    __builtin_prefetch(Arow1 + k + 32, 0, 1);
    __builtin_prefetch(Brow0 + k + 32, 0, 1);
    __builtin_prefetch(Brow1 + k + 32, 0, 1);
#pragma unroll
    for (int q = 0; q < 8; ++q) {
      int k0 = k + ((q & 4) << 2) + kb + ((q & 3) << 1);
      a0.u[q] = *(const unsigned int*)(Arow0 + k0);
      a1.u[q] = *(const unsigned int*)(Arow1 + k0);
      b0.u[q] = *(const unsigned int*)(Brow0 + k0);
      b1.u[q] = *(const unsigned int*)(Brow1 + k0);
    }
    acc00 = __builtin_amdgcn_wmma_f32_16x16x32_bf16(false, a0.v, false, b0.v,
                                                    (short)0, acc00, false, false);
    acc01 = __builtin_amdgcn_wmma_f32_16x16x32_bf16(false, a0.v, false, b1.v,
                                                    (short)0, acc01, false, false);
    acc10 = __builtin_amdgcn_wmma_f32_16x16x32_bf16(false, a1.v, false, b0.v,
                                                    (short)0, acc10, false, false);
    acc11 = __builtin_amdgcn_wmma_f32_16x16x32_bf16(false, a1.v, false, b1.v,
                                                    (short)0, acc11, false, false);
  }

  int n0 = nt + lm;
  int n1 = nt + 16 + lm;
  int mBase = mt + (lh << 3);
  if (n0 < Nreal) {
    float bv = bias ? bias[n0 / biasDiv] : 0.0f;
#pragma unroll
    for (int r = 0; r < 8; ++r) {
      C[(size_t)(mBase + r) * Np + n0]      = acc00[r] + bv;
      C[(size_t)(mBase + 16 + r) * Np + n0] = acc10[r] + bv;
    }
  }
  if (n1 < Nreal) {
    float bv = bias ? bias[n1 / biasDiv] : 0.0f;
#pragma unroll
    for (int r = 0; r < 8; ++r) {
      C[(size_t)(mBase + r) * Np + n1]      = acc01[r] + bv;
      C[(size_t)(mBase + 16 + r) * Np + n1] = acc11[r] + bv;
    }
  }
}

// ---------------------------------------------------------------------------
// im2col for 3x3 pad-1 conv over NHWC activations -> bf16 A[M][Kp]
// k = ci*9 + (dy+1)*3 + (dx+1); zero pad for k >= K.
// ---------------------------------------------------------------------------
__global__ void im2col3x3(const float* __restrict__ act, unsigned short* __restrict__ A,
                          int B_, int H, int W, int Cin, int K, int Kp)
{
  size_t idx = (size_t)blockIdx.x * blockDim.x + threadIdx.x;
  size_t total = (size_t)B_ * H * W * Kp;
  if (idx >= total) return;
  int kp = (int)(idx % Kp);
  size_t p = idx / Kp;
  float val = 0.0f;
  if (kp < K) {
    int ci = kp / 9, t = kp % 9;
    int dy = t / 3 - 1, dx = t % 3 - 1;
    int x = (int)(p % W);
    int y = (int)((p / W) % H);
    int b = (int)(p / ((size_t)W * H));
    int yy = y + dy, xx = x + dx;
    if (yy >= 0 && yy < H && xx >= 0 && xx < W)
      val = act[(((size_t)b * H + yy) * W + xx) * Cin + ci];
  }
  A[idx] = f2bf(val);
}

// activation (NHWC, C channels) -> bf16 A[P][Kp]  (for deconv GEMM, K = C)
__global__ void act_to_bf16(const float* __restrict__ act, unsigned short* __restrict__ A,
                            size_t P, int C, int Kp)
{
  size_t idx = (size_t)blockIdx.x * blockDim.x + threadIdx.x;
  size_t total = P * (size_t)Kp;
  if (idx >= total) return;
  int k = (int)(idx % Kp);
  size_t p = idx / Kp;
  A[idx] = f2bf(k < C ? act[p * C + k] : 0.0f);
}

// conv weight OIHW (O,I,3,3) -> Bt[Np][Kp], Bt[n][k] = w[n][k] (w[n] is I*9 contig)
__global__ void pack_wconv(const float* __restrict__ w, unsigned short* __restrict__ Bt,
                           int O, int K, int Np, int Kp)
{
  size_t idx = (size_t)blockIdx.x * blockDim.x + threadIdx.x;
  size_t total = (size_t)Np * Kp;
  if (idx >= total) return;
  int k = (int)(idx % Kp);
  int n = (int)(idx / Kp);
  float v = 0.0f;
  if (n < O && k < K) v = w[(size_t)n * K + k];
  Bt[idx] = f2bf(v);
}

// deconv weight (Cin, Cout, 2, 2) -> Bt[Np][Kp]; n = co*4 + (ty*2+tx); k = ci
__global__ void pack_wdeconv(const float* __restrict__ wt, unsigned short* __restrict__ Bt,
                             int Cin, int Cout, int Np, int Kp)
{
  size_t idx = (size_t)blockIdx.x * blockDim.x + threadIdx.x;
  size_t total = (size_t)Np * Kp;
  if (idx >= total) return;
  int k = (int)(idx % Kp);
  int n = (int)(idx / Kp);
  float v = 0.0f;
  if (n < Cout * 4 && k < Cin)
    v = wt[((size_t)k * Cout + (n >> 2)) * 4 + (n & 3)];
  Bt[idx] = f2bf(v);
}

// ---------------------------------------------------------------------------
// BatchNorm (training mode): per-channel mean / population variance over M rows
// ---------------------------------------------------------------------------
__global__ void bn_stats(const float* __restrict__ C, int M, int Np,
                         float* __restrict__ mean, float* __restrict__ var)
{
  int c = blockIdx.x;
  float s = 0.0f, s2 = 0.0f;
  for (int m = threadIdx.x; m < M; m += 256) {
    float v = C[(size_t)m * Np + c];
    s += v; s2 += v * v;
  }
  __shared__ float sh[256], sh2[256];
  sh[threadIdx.x] = s; sh2[threadIdx.x] = s2;
  __syncthreads();
  for (int o = 128; o > 0; o >>= 1) {
    if (threadIdx.x < o) { sh[threadIdx.x] += sh[threadIdx.x + o];
                           sh2[threadIdx.x] += sh2[threadIdx.x + o]; }
    __syncthreads();
  }
  if (threadIdx.x == 0) {
    float mu = sh[0] / (float)M;
    mean[c] = mu;
    var[c] = sh2[0] / (float)M - mu * mu;
  }
}

__global__ void bn_relu(const float* __restrict__ C,
                        const float* __restrict__ mean, const float* __restrict__ var,
                        const float* __restrict__ g, const float* __restrict__ be,
                        float* __restrict__ act, size_t M, int Nreal, int Np)
{
  size_t idx = (size_t)blockIdx.x * blockDim.x + threadIdx.x;
  size_t total = M * (size_t)Nreal;
  if (idx >= total) return;
  int c = (int)(idx % Nreal);
  size_t m = idx / Nreal;
  float v = C[m * Np + c];
  v = (v - mean[c]) * rsqrtf(var[c] + 1e-5f) * g[c] + be[c];
  act[idx] = v > 0.0f ? v : 0.0f;
}

// ---------------------------------------------------------------------------
// 2x2 max-pool, NHWC
// ---------------------------------------------------------------------------
__global__ void maxpool2_nhwc(const float* __restrict__ in, float* __restrict__ out,
                              int B_, int H, int W, int C)
{
  int Ho = H >> 1, Wo = W >> 1;
  size_t idx = (size_t)blockIdx.x * blockDim.x + threadIdx.x;
  size_t total = (size_t)B_ * Ho * Wo * C;
  if (idx >= total) return;
  int c = (int)(idx % C);
  size_t t = idx / C;
  int x = (int)(t % Wo); t /= Wo;
  int y = (int)(t % Ho);
  int b = (int)(t / Ho);
  const float* p0 = in + (((size_t)b * H + 2 * y) * W + 2 * x) * C + c;
  size_t rs = (size_t)W * C;
  float v0 = p0[0], v1 = p0[C], v2 = p0[rs], v3 = p0[rs + C];
  out[idx] = fmaxf(fmaxf(v0, v1), fmaxf(v2, v3));
}

// channel-interleave copy for concat: dst NHWC with Cd channels, write Cs at off
__global__ void chan_copy(const float* __restrict__ src, float* __restrict__ dst,
                          size_t P, int Cs, int Cd, int off)
{
  size_t idx = (size_t)blockIdx.x * blockDim.x + threadIdx.x;
  size_t total = P * (size_t)Cs;
  if (idx >= total) return;
  int c = (int)(idx % Cs);
  size_t p = idx / Cs;
  dst[p * Cd + off + c] = src[idx];
}

// scatter deconv GEMM output [P][Np] (cols co*4 + ty*2+tx) to NHWC (B,2h,2w,Cout)
__global__ void deconv_scatter(const float* __restrict__ Cg, float* __restrict__ out,
                               int B_, int h, int w, int Cout, int Np)
{
  size_t idx = (size_t)blockIdx.x * blockDim.x + threadIdx.x;
  size_t total = (size_t)B_ * h * w * Cout * 4;
  if (idx >= total) return;
  int t = (int)(idx & 3);
  int co = (int)((idx >> 2) % Cout);
  size_t p = idx / ((size_t)Cout * 4);
  int x = (int)(p % w);
  int y = (int)((p / w) % h);
  int b = (int)(p / ((size_t)w * h));
  int ty = t >> 1, tx = t & 1;
  int H2 = 2 * h, W2 = 2 * w;
  out[(((size_t)b * H2 + 2 * y + ty) * W2 + 2 * x + tx) * Cout + co] =
      Cg[p * Np + co * 4 + t];
}

// final 3x3 valid conv: (4,64,64,2) NHWC -> (4,62,62,2) NHWC, w OIHW (2,2,3,3)
__global__ void final_conv3x3(const float* __restrict__ in, const float* __restrict__ w,
                              const float* __restrict__ bias, float* __restrict__ out)
{
  int idx = blockIdx.x * blockDim.x + threadIdx.x;
  const int tot = 4 * 62 * 62 * 2;
  if (idx >= tot) return;
  int co = idx & 1;
  int t = idx >> 1;
  int x = t % 62; t /= 62;
  int y = t % 62;
  int b = t / 62;
  float s = bias[co];
#pragma unroll
  for (int ci = 0; ci < 2; ++ci)
#pragma unroll
    for (int ty = 0; ty < 3; ++ty)
#pragma unroll
      for (int tx = 0; tx < 3; ++tx)
        s += in[(((size_t)b * 64 + y + ty) * 64 + (x + tx)) * 2 + ci]
           * w[((co * 2 + ci) * 3 + ty) * 3 + tx];
  out[idx] = s;
}

// bilinear 62->64 (half-pixel centers, edge clamp); out is NCHW (4,2,64,64)
__global__ void bilinear_to_out(const float* __restrict__ f, float* __restrict__ Wout)
{
  int idx = blockIdx.x * blockDim.x + threadIdx.x;
  const int tot = 4 * 2 * 64 * 64;
  if (idx >= tot) return;
  int x = idx & 63;
  int y = (idx >> 6) & 63;
  int c = (idx >> 12) & 1;
  int b = idx >> 13;
  const float scale = 62.0f / 64.0f;
  float sy = (y + 0.5f) * scale - 0.5f;
  float sx = (x + 0.5f) * scale - 0.5f;
  float fy0 = floorf(sy), fx0 = floorf(sx);
  float wy = sy - fy0, wx = sx - fx0;
  int y0 = clampi((int)fy0, 0, 61), y1 = clampi((int)fy0 + 1, 0, 61);
  int x0 = clampi((int)fx0, 0, 61), x1 = clampi((int)fx0 + 1, 0, 61);
  auto F = [&](int yy, int xx) {
    return f[(((size_t)b * 62 + yy) * 62 + xx) * 2 + c];
  };
  Wout[idx] = (1.f - wy) * ((1.f - wx) * F(y0, x0) + wx * F(y0, x1))
            +        wy  * ((1.f - wx) * F(y1, x0) + wx * F(y1, x1));
}

// ---------------------------------------------------------------------------
// warp: out[b,x1,x2] = sum_{y1,y2} img[b,y1,y2]
//        * exp(-((x1-y1-Wx)^2+(x2-y2-Wy)^2)/1.8) / (4*pi*0.45)
// Separable: out = sum_y1 a1(y1) * ( sum_y2 img[y1][y2]*a2(y2) ).
// ---------------------------------------------------------------------------
__global__ void __launch_bounds__(256)
warp_kernel(const float* __restrict__ x, const float* __restrict__ Wout,
            float* __restrict__ ypred)
{
  __shared__ float img[64 * 64];
  int b = blockIdx.y;
  for (int i = threadIdx.x; i < 4096; i += 256) img[i] = x[(size_t)b * 4096 + i];
  __syncthreads();

  int pix = blockIdx.x * 256 + threadIdx.x;   // 0..4095
  int x1 = pix >> 6, x2 = pix & 63;
  float Wx = Wout[((size_t)b * 2 + 0) * 4096 + pix];
  float Wy = Wout[((size_t)b * 2 + 1) * 4096 + pix];
  const float inv_c = 1.0f / 1.8f;            // 1/(4*D_COEF*DT)

  float a2[64];
#pragma unroll
  for (int y2 = 0; y2 < 64; ++y2) {
    float d = (float)(x2 - y2) - Wy;
    a2[y2] = __expf(-d * d * inv_c);
  }
  float s = 0.0f;
  for (int y1 = 0; y1 < 64; ++y1) {
    float d = (float)(x1 - y1) - Wx;
    float a1 = __expf(-d * d * inv_c);
    float inner = 0.0f;
#pragma unroll
    for (int y2 = 0; y2 < 64; ++y2) inner += img[y1 * 64 + y2] * a2[y2];
    s += a1 * inner;
  }
  ypred[(size_t)b * 4096 + pix] = s * (1.0f / (4.0f * 3.14159265358979323846f * 0.45f));
}

// ---------------------------------------------------------------------------
// Host orchestration
// ---------------------------------------------------------------------------
static inline size_t cdiv_sz(size_t a, size_t b) { return (a + b - 1) / b; }
static inline int alignup(int v, int a) { return (v + a - 1) / a * a; }

extern "C" void kernel_launch(void* const* d_in, const int* in_sizes, int n_in,
                              void* d_out, int out_size, void* d_ws, size_t ws_size,
                              hipStream_t stream)
{
  (void)in_sizes; (void)n_in; (void)out_size; (void)ws_size;

  // JAX tree-flatten (alphabetical) leaf order assumption:
  //   params < x ; dec1,dec2,dec3,dec4,enc1..enc4,final ; within layer keys sorted.
  // dec*: b1,b2,be1,be2,bt,g1,g2,w1,w2,wt  (10 leaves)
  // enc*: b1,b2,be1,be2,g1,g2,w1,w2        (8 leaves)
  // final: b,w ; then x.
  const int DEC1 = 0, DEC2 = 10, DEC3 = 20, DEC4 = 30;
  const int ENC1 = 40, ENC2 = 48, ENC3 = 56, ENC4 = 64;
  const int FIN_B = 72, FIN_W = 73, XIN = 74;
  auto F = [&](int i) { return (const float*)d_in[i]; };

  // workspace layout
  size_t off = 0;
  auto alloc = [&](size_t bytes) {
    void* p = (char*)d_ws + off;
    off += (bytes + 255) & ~(size_t)255;
    return p;
  };
  unsigned short* A   = (unsigned short*)alloc((size_t)20 << 20);
  unsigned short* Bt  = (unsigned short*)alloc((size_t)5 << 20);
  float* Cg   = (float*)alloc((size_t)5 << 20);
  float* meanB = (float*)alloc(2048 * 4);
  float* varB  = (float*)alloc(2048 * 4);
  float* tmp1 = (float*)alloc((size_t)4 << 20);   // dc intermediate
  float* actB = (float*)alloc((size_t)4 << 20);   // dc output
  float* s1   = (float*)alloc((size_t)1 << 20);   // (4,32,32,64)
  float* s2   = (float*)alloc((size_t)512 << 10); // (4,16,16,128)
  float* s3   = (float*)alloc((size_t)256 << 10); // (4,8,8,256)
  float* e4   = (float*)alloc(131072);            // (4,4,4,512)
  float* dbuf = (float*)alloc((size_t)2 << 20);
  float* catb = (float*)alloc((size_t)3 << 20);
  float* fbuf = (float*)alloc((size_t)4 * 62 * 62 * 2 * 4);

  auto launch_gemm = [&](int M, int Nreal, int Np, int Kp,
                         const float* bias, int biasDiv) {
    int blocks32 = (M / 32) * (Np / 32);          // one 32x32 C block per wave
    dim3 blk(32, 8);
    wmma_gemm_bf16<<<(unsigned)cdiv_sz(blocks32, 8), blk, 0, stream>>>(
        A, Bt, bias, biasDiv, Cg, M, Nreal, Np, Kp);
  };

  // conv3x3(pad1) + BN(train) + ReLU  : NHWC in -> NHWC out
  auto conv_bn_relu = [&](const float* inAct, const float* w,
                          const float* g, const float* be,
                          int B_, int H, int W, int Cin, int Cout, float* outAct) {
    int M = B_ * H * W;
    int K = Cin * 9, Kp = alignup(K, 32);
    int Np = alignup(Cout, 32);
    im2col3x3<<<(unsigned)cdiv_sz((size_t)M * Kp, 256), 256, 0, stream>>>(
        inAct, A, B_, H, W, Cin, K, Kp);
    pack_wconv<<<(unsigned)cdiv_sz((size_t)Np * Kp, 256), 256, 0, stream>>>(
        w, Bt, Cout, K, Np, Kp);
    launch_gemm(M, Cout, Np, Kp, nullptr, 1);     // conv bias cancelled by BN
    bn_stats<<<Cout, 256, 0, stream>>>(Cg, M, Np, meanB, varB);
    bn_relu<<<(unsigned)cdiv_sz((size_t)M * Cout, 256), 256, 0, stream>>>(
        Cg, meanB, varB, g, be, outAct, (size_t)M, Cout, Np);
  };

  struct DCP { const float *w1, *g1, *be1, *w2, *g2, *be2; };
  auto enc_p = [&](int base) {
    return DCP{F(base + 6), F(base + 4), F(base + 2), F(base + 7), F(base + 5), F(base + 3)};
  };
  auto dec_p = [&](int base) {
    return DCP{F(base + 7), F(base + 5), F(base + 2), F(base + 8), F(base + 6), F(base + 3)};
  };
  auto run_dc = [&](const float* inAct, int B_, int H, int W,
                    int cin, int cmid, DCP p, float* outAct) {
    conv_bn_relu(inAct, p.w1, p.g1, p.be1, B_, H, W, cin, cmid, tmp1);
    conv_bn_relu(tmp1, p.w2, p.g2, p.be2, B_, H, W, cmid, cmid, outAct);
  };
  auto run_pool = [&](const float* in, float* out, int B_, int H, int W, int C) {
    size_t tot = (size_t)B_ * (H / 2) * (W / 2) * C;
    maxpool2_nhwc<<<(unsigned)cdiv_sz(tot, 256), 256, 0, stream>>>(in, out, B_, H, W, C);
  };
  auto run_deconv = [&](const float* inAct, const float* wt, const float* bt,
                        int B_, int h, int w, int Cin, int Cout, float* outAct) {
    int M = B_ * h * w;
    int Kp = alignup(Cin, 32);
    int N = Cout * 4, Np = alignup(N, 32);
    act_to_bf16<<<(unsigned)cdiv_sz((size_t)M * Kp, 256), 256, 0, stream>>>(
        inAct, A, (size_t)M, Cin, Kp);
    pack_wdeconv<<<(unsigned)cdiv_sz((size_t)Np * Kp, 256), 256, 0, stream>>>(
        wt, Bt, Cin, Cout, Np, Kp);
    launch_gemm(M, N, Np, Kp, bt, 4);
    deconv_scatter<<<(unsigned)cdiv_sz((size_t)M * Cout * 4, 256), 256, 0, stream>>>(
        Cg, outAct, B_, h, w, Cout, Np);
  };
  auto run_cat = [&](const float* a, int Ca, const float* b, int Cb,
                     float* dst, size_t P) {
    int Cd = Ca + Cb;
    chan_copy<<<(unsigned)cdiv_sz(P * Ca, 256), 256, 0, stream>>>(a, dst, P, Ca, Cd, 0);
    chan_copy<<<(unsigned)cdiv_sz(P * Cb, 256), 256, 0, stream>>>(b, dst, P, Cb, Cd, Ca);
  };

  const float* X = F(XIN);                    // (4,1,64,64) NCHW == NHWC (C=1)

  // ---- Encoder ----
  run_dc(X, 4, 64, 64, 1, 64, enc_p(ENC1), actB);
  run_pool(actB, s1, 4, 64, 64, 64);                  // (4,32,32,64)
  run_dc(s1, 4, 32, 32, 64, 128, enc_p(ENC2), actB);
  run_pool(actB, s2, 4, 32, 32, 128);                 // (4,16,16,128)
  run_dc(s2, 4, 16, 16, 128, 256, enc_p(ENC3), actB);
  run_pool(actB, s3, 4, 16, 16, 256);                 // (4,8,8,256)
  run_dc(s3, 4, 8, 8, 256, 512, enc_p(ENC4), actB);
  run_pool(actB, e4, 4, 8, 8, 512);                   // (4,4,4,512)

  // ---- Decoder ----
  run_dc(e4, 4, 4, 4, 512, 512, dec_p(DEC4), actB);
  run_deconv(actB, F(DEC4 + 9), F(DEC4 + 4), 4, 4, 4, 512, 386, dbuf);   // (4,8,8,386)

  run_cat(dbuf, 386, s3, 256, catb, (size_t)4 * 8 * 8);
  run_dc(catb, 4, 8, 8, 642, 256, dec_p(DEC3), actB);
  run_deconv(actB, F(DEC3 + 9), F(DEC3 + 4), 4, 8, 8, 256, 194, dbuf);   // (4,16,16,194)

  run_cat(dbuf, 194, s2, 128, catb, (size_t)4 * 16 * 16);
  run_dc(catb, 4, 16, 16, 322, 128, dec_p(DEC2), actB);
  run_deconv(actB, F(DEC2 + 9), F(DEC2 + 4), 4, 16, 16, 128, 98, dbuf);  // (4,32,32,98)

  run_cat(dbuf, 98, s1, 64, catb, (size_t)4 * 32 * 32);
  run_dc(catb, 4, 32, 32, 162, 64, dec_p(DEC1), actB);
  run_deconv(actB, F(DEC1 + 9), F(DEC1 + 4), 4, 32, 32, 64, 2, dbuf);    // (4,64,64,2)

  // ---- Final conv + resize -> W (NCHW) ----
  float* Wout = (float*)d_out;                // 4*2*64*64 = 32768 floats
  final_conv3x3<<<(unsigned)cdiv_sz((size_t)4 * 62 * 62 * 2, 256), 256, 0, stream>>>(
      dbuf, F(FIN_W), F(FIN_B), fbuf);
  bilinear_to_out<<<(unsigned)cdiv_sz((size_t)4 * 2 * 64 * 64, 256), 256, 0, stream>>>(
      fbuf, Wout);

  // ---- Warp -> y_pred ----
  float* ypred = Wout + 4 * 2 * 64 * 64;      // next 4*64*64 floats
  warp_kernel<<<dim3(16, 4), 256, 0, stream>>>(X, Wout, ypred);
}
